// Net_73787538145697
// MI455X (gfx1250) — compile-verified, matrix operand order checked by reading, not run on hardware
//
#include <hip/hip_runtime.h>

#define N0v 409600
#define N1v 40960
#define N2v 4096
#define E1v 409600
#define E2v 40960
#define FINv 512
#define FHIDv 256
#define FOUTv 128

typedef __attribute__((ext_vector_type(16))) __bf16 v16bf;
typedef __attribute__((ext_vector_type(8)))  __bf16 v8bf;
typedef __attribute__((ext_vector_type(8)))  float  v8f;

__device__ __forceinline__ unsigned short f2bfbits(float f) {
  unsigned u = __builtin_bit_cast(unsigned, f);
  u += 0x7fffu + ((u >> 16) & 1u);           // round-to-nearest-even
  return (unsigned short)(u >> 16);
}
__device__ __forceinline__ __bf16 f2bf(float f) {
  unsigned short s = f2bfbits(f);
  return __builtin_bit_cast(__bf16, s);
}

// --- A fragment: 16 bf16 elements for V_WMMA_F32_16X16X32_BF16 ---
// ISA 7.12.2: lane L: M = L%16, h = L/16. Elements e<8 -> K = c0+e,
// e>=8 -> K = c0+16+(e-8), where c0 = kb + 8h. Two contiguous 32B f32 loads,
// converted (and mean-scaled) in registers; cvt VALU co-executes with XDL WMMA.
__device__ __forceinline__ v16bf load_a_frag(const float* __restrict__ row,
                                             int c0, float scale) {
  const float4* p0 = reinterpret_cast<const float4*>(row + c0);
  const float4* p1 = reinterpret_cast<const float4*>(row + c0 + 16);
  float4 f0 = p0[0], f1 = p0[1], f2 = p1[0], f3 = p1[1];
  v16bf a;
  a[0]  = f2bf(f0.x * scale); a[1]  = f2bf(f0.y * scale);
  a[2]  = f2bf(f0.z * scale); a[3]  = f2bf(f0.w * scale);
  a[4]  = f2bf(f1.x * scale); a[5]  = f2bf(f1.y * scale);
  a[6]  = f2bf(f1.z * scale); a[7]  = f2bf(f1.w * scale);
  a[8]  = f2bf(f2.x * scale); a[9]  = f2bf(f2.y * scale);
  a[10] = f2bf(f2.z * scale); a[11] = f2bf(f2.w * scale);
  a[12] = f2bf(f3.x * scale); a[13] = f2bf(f3.y * scale);
  a[14] = f2bf(f3.z * scale); a[15] = f2bf(f3.w * scale);
  return a;
}

// --- B fragment: weights pre-transposed to N-major bf16 (row n = K-run) ---
// Same K-pattern as A with N = lane%16; two contiguous 16B bf16 loads.
__device__ __forceinline__ v16bf load_b_frag(const unsigned short* __restrict__ row,
                                             int c0) {
  v8bf lo = *reinterpret_cast<const v8bf*>(row + c0);
  v8bf hi = *reinterpret_cast<const v8bf*>(row + c0 + 16);
  v16bf b;
#pragma unroll
  for (int i = 0; i < 8; ++i) { b[i] = lo[i]; b[8 + i] = hi[i]; }
  return b;
}

// Pre-transpose & convert weights: Bt[n][k] = bf16( k<Khalf ? Wl[k][n] : Wr[k-Khalf][n] )
__global__ void build_bt(const float* __restrict__ Wl, const float* __restrict__ Wr,
                         unsigned short* __restrict__ Bt, int Ktot, int Khalf, int N) {
  int k = blockIdx.x;
  int n = threadIdx.x;
  float w = (k < Khalf) ? Wl[(size_t)k * N + n] : Wr[(size_t)(k - Khalf) * N + n];
  Bt[(size_t)n * Ktot + k] = f2bfbits(w);
}

// Edge scatter: agg[dst] += x[src] (per-element f32 atomics resolve in L2:
// agg1 is 84 MB < 192 MB global L2), cnt[dst] += 1. Bandwidth-roofline phase.
__global__ void scatter_accum(const float* __restrict__ xsrc,
                              const int* __restrict__ src, const int* __restrict__ dst,
                              float* __restrict__ agg, float* __restrict__ cnt,
                              int F4, int ld) {
  int e = blockIdx.x;
  int s = src[e], d = dst[e];
  const float4* xr = reinterpret_cast<const float4*>(xsrc + (size_t)s * ld);
  float* ar = agg + (size_t)d * ld;
  int t = threadIdx.x;
  if (t < F4) {
    float4 v = xr[t];
    atomicAdd(ar + t * 4 + 0, v.x);
    atomicAdd(ar + t * 4 + 1, v.y);
    atomicAdd(ar + t * 4 + 2, v.z);
    atomicAdd(ar + t * 4 + 3, v.w);
  }
  if (t == 0) atomicAdd(cnt + d, 1.0f);
}

// Fused SAGE layer: out = act( [agg*inv_cnt | xdst] @ Bt^T + bias )
// One wave computes a 16 x (16*NSUP) output tile: one A fragment feeds NSUP
// WMMAs per K-step (A traffic & cvt VALU amortized NSUP-fold).
template <int NSUP>
__global__ __launch_bounds__(256)
void sage_gemm_wmma(const float* __restrict__ agg, const float* __restrict__ cnt,
                    const float* __restrict__ xdst, const unsigned short* __restrict__ Bt,
                    const float* __restrict__ bias, float* __restrict__ out,
                    int M, int N, int Khalf, int do_relu) {
  const int lane = threadIdx.x & 31;
  const int wave = threadIdx.x >> 5;
  const int tile = blockIdx.x * (blockDim.x >> 5) + wave;
  const int ntn  = N / (16 * NSUP);
  const int mt   = tile / ntn;
  const int nsup = tile - mt * ntn;
  if (mt * 16 >= M) return;                 // uniform per wave (grids are exact)

  const int l16  = lane & 15;
  const int hgrp = lane >> 4;
  const int m    = mt * 16 + l16;
  const int n0   = nsup * (16 * NSUP) + l16;
  const int Ktot = 2 * Khalf;

  const float inv = 1.0f / fmaxf(cnt[m], 1.0f);
  const float* arow = agg  + (size_t)m * Khalf;
  const float* xrow = xdst + (size_t)m * Khalf;

  const unsigned short* brow[NSUP];
#pragma unroll
  for (int j = 0; j < NSUP; ++j)
    brow[j] = Bt + (size_t)(n0 + 16 * j) * Ktot;

  v8f c[NSUP];
#pragma unroll
  for (int j = 0; j < NSUP; ++j) c[j] = (v8f){};

  // first K-half: mean = agg * inv_cnt
  for (int kb = 0; kb < Khalf; kb += 32) {
    const int c0 = kb + hgrp * 8;
    v16bf a = load_a_frag(arow, c0, inv);
#pragma unroll
    for (int j = 0; j < NSUP; ++j) {
      v16bf b = load_b_frag(brow[j], c0);
      c[j] = __builtin_amdgcn_wmma_f32_16x16x32_bf16(false, a, false, b,
                                                     (short)0, c[j], false, false);
    }
  }
  // second K-half: x_dst (unscaled), B offset by Khalf
  for (int kb = 0; kb < Khalf; kb += 32) {
    const int c0 = kb + hgrp * 8;
    v16bf a = load_a_frag(xrow, c0, 1.0f);
#pragma unroll
    for (int j = 0; j < NSUP; ++j) {
      v16bf b = load_b_frag(brow[j] + Khalf, c0);
      c[j] = __builtin_amdgcn_wmma_f32_16x16x32_bf16(false, a, false, b,
                                                     (short)0, c[j], false, false);
    }
  }

  // C/D layout (ISA 7.12.2): VGPR r -> row = mt*16 + 8*hgrp + r, col = n0+16j
#pragma unroll
  for (int j = 0; j < NSUP; ++j) {
    const int col = n0 + 16 * j;
    const float bv = bias[col];
#pragma unroll
    for (int r = 0; r < 8; ++r) {
      int row = mt * 16 + hgrp * 8 + r;
      float v = c[j][r] + bv;
      if (do_relu) v = fmaxf(v, 0.0f);
      out[(size_t)row * N + col] = v;
    }
  }
}

extern "C" void kernel_launch(void* const* d_in, const int* in_sizes, int n_in,
                              void* d_out, int out_size, void* d_ws, size_t ws_size,
                              hipStream_t stream) {
  const float* x   = (const float*)d_in[0];
  const int*   e1s = (const int*)d_in[1];
  const int*   e1d = (const int*)d_in[2];
  const int*   e2s = (const int*)d_in[3];
  const int*   e2d = (const int*)d_in[4];
  const float* W1l = (const float*)d_in[5];
  const float* b1  = (const float*)d_in[6];
  const float* W1r = (const float*)d_in[7];
  const float* W2l = (const float*)d_in[8];
  const float* b2  = (const float*)d_in[9];
  const float* W2r = (const float*)d_in[10];
  float* out = (float*)d_out;

  // workspace layout (zeroed region first, contiguous -> single memset)
  char* w = (char*)d_ws;
  size_t off = 0;
  auto take = [&](size_t bytes) -> void* {
    void* p = w + off;
    off += (bytes + 255) & ~(size_t)255;
    return p;
  };
  float* agg1 = (float*)take((size_t)N1v * FINv * 4);   // 83.9 MB
  float* cnt1 = (float*)take((size_t)N1v * 4);
  float* agg2 = (float*)take((size_t)N2v * FHIDv * 4);  // 4.2 MB
  float* cnt2 = (float*)take((size_t)N2v * 4);
  const size_t zero_bytes = off;
  float*          h   = (float*)take((size_t)N1v * FHIDv * 4);        // 41.9 MB
  unsigned short* Bt1 = (unsigned short*)take((size_t)FHIDv * 2 * FINv * 2);  // 256x1024 bf16
  unsigned short* Bt2 = (unsigned short*)take((size_t)FOUTv * 2 * FHIDv * 2); // 128x512 bf16

  hipMemsetAsync(d_ws, 0, zero_bytes, stream);

  // weight transpose + bf16 convert (tiny, re-done every call for determinism)
  build_bt<<<2 * FINv,  FHIDv, 0, stream>>>(W1l, W1r, Bt1, 2 * FINv,  FINv,  FHIDv);
  build_bt<<<2 * FHIDv, FOUTv, 0, stream>>>(W2l, W2r, Bt2, 2 * FHIDv, FHIDv, FOUTv);

  // layer 1: M=40960, N=256, K=512+512, NSUP=4 -> 2560*4 = 10240 waves = 1280 blocks
  scatter_accum<<<E1v, 128, 0, stream>>>(x, e1s, e1d, agg1, cnt1, FINv / 4, FINv);
  {
    const int tiles = (N1v / 16) * (FHIDv / 64);
    sage_gemm_wmma<4><<<tiles / 8, 256, 0, stream>>>(agg1, cnt1, x, Bt1, b1, h,
                                                     N1v, FHIDv, FINv, /*relu=*/1);
  }
  // layer 2: M=4096, N=128, K=256+256, NSUP=2 -> 256*4 = 1024 waves = 128 blocks
  scatter_accum<<<E2v, 64, 0, stream>>>(h, e2s, e2d, agg2, cnt2, FHIDv / 4, FHIDv);
  {
    const int tiles = (N2v / 16) * (FOUTv / 32);
    sage_gemm_wmma<2><<<tiles / 8, 256, 0, stream>>>(agg2, cnt2, h, Bt2, b2, out,
                                                     N2v, FOUTv, FHIDv, /*relu=*/0);
  }
}
